// Bio_All_Network_66039417143712
// MI455X (gfx1250) — matrix operand order repaired and checked by reading, not compiled, not run on hardware
//
#include <hip/hip_runtime.h>
#include <hip/hip_bf16.h>
#include <math.h>

typedef __attribute__((ext_vector_type(8)))  float   v8f;
typedef __attribute__((ext_vector_type(8)))  __bf16  v8bf;
typedef __attribute__((ext_vector_type(16))) __bf16  v16bf;

#define NATOMS 128
#define BZ 2
#define KH 96   // padded K for h: 64 hidden + 1 bias-row(=1.0) + 31 zeros (3x K=32 WMMA steps)
#define INV_SQRTN 0.08838834764831845f
#define INV_SQRT3 0.5773502691896258f
#define INV_SQRT2 0.7071067811865476f

__device__ __forceinline__ float softplus5(float x){ float y = 5.0f*x; return fmaxf(y,0.0f) + log1pf(__expf(-fabsf(y))); }
__device__ __forceinline__ float ssp_act(float x){ return (softplus5(x) - 0.69314718056f)*0.2f; }  // shifted softplus
__device__ __forceinline__ float sp_act(float x){ return softplus5(x)*0.2f; }
__device__ __forceinline__ float sigmoidf_(float x){ return 1.0f/(1.0f+__expf(-x)); }

__device__ __forceinline__ v8bf ldbf8(const __bf16* p){ return *reinterpret_cast<const v8bf*>(p); }

__device__ __forceinline__ v16bf pack16(v8bf lo, v8bf hi){
  v16bf r;
#pragma unroll
  for(int j=0;j<8;++j){ r[j]=lo[j]; r[8+j]=hi[j]; }
  return r;
}

__device__ __forceinline__ v8f wmma_bf16(v16bf a, v16bf b, v8f c){
  // D = A(16x32 bf16) * B(32x16 bf16) + C(16x16 f32)
  return __builtin_amdgcn_wmma_f32_16x16x32_bf16(false, a, false, b, (short)0, c, false, false);
}

// ---------------- util: zero a float buffer --------------------------------------------
__global__ void zero_kernel(float* __restrict__ p, int n){
  int i = blockIdx.x*blockDim.x + threadIdx.x;
  if (i < n) p[i] = 0.0f;
}

// ---------------- K0: encode features -> g0 for layer1 (mask & 1/sqrt(N) folded) ---------
__global__ void encode_kernel(const float* __restrict__ features, const float* __restrict__ mask,
                              const float* __restrict__ Wb, const float* __restrict__ bb,
                              const float* __restrict__ Wc, const float* __restrict__ bc,
                              float* __restrict__ g0l1){
  int t = blockIdx.x*blockDim.x + threadIdx.x;
  if (t >= BZ*NATOMS) return;
  float mk = mask[t] * INV_SQRTN;
  const float* fr = features + t*8;
#pragma unroll
  for(int u=0;u<16;++u){
    float sb = bb[u];
#pragma unroll
    for(int v=0;v<7;++v) sb += fr[v]*Wb[v*16+u];
    float sc = bc[u] + fr[7]*Wc[u];
    g0l1[(t*2+0)*16+u] = sb*mk;
    g0l1[(t*2+1)*16+u] = sc*mk;
  }
}

// ---------------- K1: pairwise geometry: Y1 (f32) + cosine basis (bf16, padded to 16) ----
__global__ void geom_kernel(const float* __restrict__ geometry,
                            __bf16* __restrict__ basis, float* __restrict__ Y1){
  int idx = blockIdx.x*blockDim.x + threadIdx.x;   // (z, a, b)
  if (idx >= BZ*NATOMS*NATOMS) return;
  int b = idx % NATOMS; int a = (idx/NATOMS)%NATOMS; int z = idx/(NATOMS*NATOMS);
  const float* ga = geometry + (z*NATOMS+a)*3;
  const float* gb = geometry + (z*NATOMS+b)*3;
  float d0 = gb[0]-ga[0], d1 = gb[1]-ga[1], d2 = gb[2]-ga[2];
  float sq = d0*d0+d1*d1+d2*d2;
  float r, rinv;
  if (sq > 1e-24f){ r = sqrtf(sq); rinv = 1.0f/r; } else { r = 0.0f; rinv = 0.0f; }
  Y1[(size_t)idx*3+0]=d0*rinv; Y1[(size_t)idx*3+1]=d1*rinv; Y1[(size_t)idx*3+2]=d2*rinv;
  const float step = 10.0f/9.0f;
  __bf16* bp = basis + (size_t)idx*16;
#pragma unroll
  for(int tt=0;tt<10;++tt){
    float dd = (r - step*(float)tt)/step;
    float val = 0.0f;
    if (fabsf(dd) < 1.0f){ float cs = __cosf(1.5707963267948966f*dd); val = cs*cs; }
    bp[tt] = (__bf16)val;
  }
#pragma unroll
  for(int tt=10;tt<16;++tt) bp[tt] = (__bf16)0.0f;
}

// ---------------- K2: radial MLP, WMMA. 4 waves/block, 1 tile of 16 pairs per wave -------
__global__ void radial_kernel(const __bf16* __restrict__ basis,
                              const float* __restrict__ W0, const float* __restrict__ b0,
                              const float* __restrict__ W1, const float* __restrict__ b1,
                              __bf16* __restrict__ hout){
  __shared__ __align__(16) __bf16 W0T[64][32];     // [n][k], k padded 10->32
  __shared__ __align__(16) __bf16 W1T[64][64];     // [n][k]
  __shared__ float b0s[64], b1s[64];
  __shared__ __align__(16) __bf16 h0buf[4][16][64];  // per-wave activated hidden tile [m][n]
  int tid = threadIdx.x;
  for(int i=tid;i<64*32;i+=blockDim.x){ int n=i/32,k=i%32; W0T[n][k] = (k<10)?(__bf16)W0[k*64+n]:(__bf16)0.0f; }
  for(int i=tid;i<64*64;i+=blockDim.x){ int n=i/64,k=i%64; W1T[n][k] = (__bf16)W1[k*64+n]; }
  if (tid<64){ b0s[tid]=b0[tid]; b1s[tid]=b1[tid]; }
  __syncthreads();

  int wave = tid>>5, lane = tid&31;
  int tile = blockIdx.x*4 + wave;       // 2048 tiles total
  int P0 = tile*16;
  int mrow = lane & 15, khalf = lane>>4;

  // A0: 16x32 basis tile (K 16..31 zero). elems 0..7 <- K=khalf*8+j ; 8..15 <- K>=16 (zero)
  v8bf z8 = {};
  v16bf A0 = pack16(ldbf8(basis + (size_t)(P0+mrow)*16 + khalf*8), z8);

#pragma unroll
  for(int nt=0;nt<4;++nt){
    v8f acc = {};
    const __bf16* wp = &W0T[nt*16 + mrow][khalf*8];
    acc = wmma_bf16(A0, pack16(ldbf8(wp), ldbf8(wp+16)), acc);
    int col = nt*16 + mrow;
    int rb = khalf*8;
#pragma unroll
    for(int i=0;i<8;++i){
      float v = ssp_act(acc[i] + b0s[col]);
      h0buf[wave][rb+i][col] = (__bf16)v;
    }
  }
  __syncthreads();

  v16bf A1[2];
#pragma unroll
  for(int ks=0;ks<2;++ks){
    const __bf16* p = &h0buf[wave][mrow][ks*32 + khalf*8];
    A1[ks] = pack16(ldbf8(p), ldbf8(p+16));
  }
#pragma unroll
  for(int nt=0;nt<4;++nt){
    v8f acc = {};
#pragma unroll
    for(int ks=0;ks<2;++ks){
      const __bf16* p = &W1T[nt*16 + mrow][ks*32 + khalf*8];
      acc = wmma_bf16(A1[ks], pack16(ldbf8(p), ldbf8(p+16)), acc);
    }
    int col = nt*16 + mrow;
    int rb = khalf*8;
#pragma unroll
    for(int i=0;i<8;++i){
      float v = ssp_act(acc[i] + b1s[col]);
      hout[(size_t)(P0+rb+i)*KH + col] = (__bf16)v;
    }
  }
  // constant row k=64 (carries folded biases) and zero pad k=65..95
  if (lane < 16){
    __bf16* p = hout + (size_t)(P0+lane)*KH + 64;
    p[0] = (__bf16)1.0f;
#pragma unroll
    for(int t=1;t<32;++t) p[t] = (__bf16)0.0f;
  }
}

// ---------------- K3a: build per-b effective matrix E1 [z][b][n=96][k=96] -----------------
__global__ void e1_build_kernel(const float* __restrict__ g0l1,
                                const float* __restrict__ Wout, const float* __restrict__ bout,
                                __bf16* __restrict__ E1){
  int idx = blockIdx.x*blockDim.x + threadIdx.x;   // == (z*128+b)*96 + n
  if (idx >= BZ*NATOMS*96) return;
  int n = idx % 96; int zb = idx/96;
  int br = n/48, np = n%48;
  const float* g = g0l1 + (zb*2 + br)*16;
  int colbase = (np<32) ? np*16 : 512 + (np-32)*16;
  __bf16* E = E1 + (size_t)idx*KH;
  for(int k=0;k<64;++k){
    const float* wr = Wout + (size_t)k*768 + colbase;
    float s = 0.f;
#pragma unroll
    for(int v=0;v<16;++v) s += wr[v]*g[v];
    E[k] = (__bf16)s;
  }
  { const float* wr = bout + colbase; float s=0.f;
#pragma unroll
    for(int v=0;v<16;++v) s += wr[v]*g[v];
    E[64]=(__bf16)s; }
#pragma unroll
  for(int k=65;k<KH;++k) E[k]=(__bf16)0.0f;
}

// ---------------- K4a: build E2 [z][b][n=480][k=96] --------------------------------------
__global__ void e2_build_kernel(const float* __restrict__ g0l2, const float* __restrict__ g1l2,
                                const float* __restrict__ Wout, const float* __restrict__ bout,
                                __bf16* __restrict__ E2){
  int idx = blockIdx.x*blockDim.x + threadIdx.x;   // == (z*128+b)*480 + n
  if (idx >= BZ*NATOMS*480) return;
  int n = idx % 480; int zb = idx/480;
  int br = n/240, np = n%240;
  const float* g0 = g0l2 + (zb*2+br)*16;
  const float* g1 = g1l2 + (zb*2+br)*48;   // [v][m]
  float gv[16]; float scale = 1.0f; int colbase;
  if (np < 32){ colbase = np*16; for(int v=0;v<16;++v) gv[v]=g0[v]; }
  else if (np < 48){ colbase = 512 + (np-32)*16; for(int v=0;v<16;++v) gv[v]=g0[v]; }
  else if (np < 144){ int q=np-48; int u=q/3, m=q%3; colbase = 768 + u*16;
                      for(int v=0;v<16;++v) gv[v]=g1[v*3+m]; scale=INV_SQRT3; }
  else if (np < 192){ int q=np-144; int u=q/3, m=q%3; colbase = 1280 + u*16;
                      for(int v=0;v<16;++v) gv[v]=g1[v*3+m]; }
  else              { int q=np-192; int u=q/3, ii=q%3; colbase = 1536 + u*16;
                      for(int v=0;v<16;++v) gv[v]=g1[v*3+ii]; scale=INV_SQRT2; }
  __bf16* E = E2 + (size_t)idx*KH;
  for(int k=0;k<64;++k){
    const float* wr = Wout + (size_t)k*1792 + colbase;
    float s = 0.f;
#pragma unroll
    for(int v=0;v<16;++v) s += wr[v]*gv[v];
    E[k]=(__bf16)(s*scale);
  }
  { const float* wr = bout + colbase; float s=0.f;
#pragma unroll
    for(int v=0;v<16;++v) s += wr[v]*gv[v];
    E[64]=(__bf16)(s*scale); }
#pragma unroll
  for(int k=65;k<KH;++k) E[k]=(__bf16)0.0f;
}

// ---------------- K3b/K4b: pair contraction T = h @ E_b (WMMA), b-split across blocks ----
// grid: z(2) x atile(8) x bchunk(8) = 128 blocks, 8 waves each, 2 b's per wave.
// Partial o0/o1 accumulated in LDS, flushed once per block to global with atomics.
template<int NC, int LAYER>
__global__ void contract_accum_kernel(const __bf16* __restrict__ h, const __bf16* __restrict__ E,
                                      const float* __restrict__ Y1,
                                      float* __restrict__ o0g, float* __restrict__ o1g){
  __shared__ float o0s[2][16][32];
  __shared__ float o1s[2][16][16][3];
  int tid = threadIdx.x;
  for(int i=tid;i<2*16*32;i+=256) ((float*)o0s)[i]=0.f;
  for(int i=tid;i<2*16*16*3;i+=256) ((float*)o1s)[i]=0.f;
  __syncthreads();

  int z  = blockIdx.x >> 6;
  int a0 = ((blockIdx.x >> 3) & 7)*16;
  int bc = blockIdx.x & 7;
  int wave = tid>>5, lane = tid&31;
  int nlo = lane & 15, khalf = lane>>4;
  int rb = khalf*8;                  // D-matrix row base for this lane-half
  const int NT = NC/16;

  for(int bi=wave; bi<16; bi+=8){
    int b = bc*16 + bi;
    // prefetch the next-b streams into the cache hierarchy (global_prefetch_b8)
    if (bi+8 < 16){
      __builtin_prefetch(E + (size_t)(z*NATOMS+b+8)*NC*KH, 0, 1);
      __builtin_prefetch(h + ((size_t)((z*NATOMS + a0 + nlo)*NATOMS + b+8))*KH, 0, 1);
    }
    float Yv[8][3];
#pragma unroll
    for(int i=0;i<8;++i){
      const float* yp = Y1 + ((size_t)((z*NATOMS + a0 + rb + i)*NATOMS + b))*3;
      Yv[i][0]=yp[0]; Yv[i][1]=yp[1]; Yv[i][2]=yp[2];
    }
    v16bf A[3];
#pragma unroll
    for(int ks=0;ks<3;++ks){
      const __bf16* p = h + ((size_t)((z*NATOMS + a0 + nlo)*NATOMS + b))*KH + ks*32 + khalf*8;
      A[ks] = pack16(ldbf8(p), ldbf8(p+16));
    }
    const __bf16* Eb = E + (size_t)(z*NATOMS+b)*NC*KH;
#pragma unroll
    for(int nt=0; nt<NT; ++nt){
      v8f acc = {};
#pragma unroll
      for(int ks=0;ks<3;++ks){
        const __bf16* p = Eb + (size_t)(nt*16 + nlo)*KH + ks*32 + khalf*8;
        acc = wmma_bf16(A[ks], pack16(ldbf8(p), ldbf8(p+16)), acc);
      }
      int c = nt*16 + nlo;
      if (LAYER == 1){
        int br = c/48, n = c%48;
        if (n < 32){
#pragma unroll
          for(int i=0;i<8;++i) atomicAdd(&o0s[br][rb+i][n], acc[i]);
        } else {
          int u = n-32;
#pragma unroll
          for(int i=0;i<8;++i){
            atomicAdd(&o1s[br][rb+i][u][0], acc[i]*Yv[i][0]);
            atomicAdd(&o1s[br][rb+i][u][1], acc[i]*Yv[i][1]);
            atomicAdd(&o1s[br][rb+i][u][2], acc[i]*Yv[i][2]);
          }
        }
      } else {
        int br = c/240, n = c%240;
        if (n < 32){
#pragma unroll
          for(int i=0;i<8;++i) atomicAdd(&o0s[br][rb+i][n], acc[i]);
        } else if (n < 48){
          int u = n-32;
#pragma unroll
          for(int i=0;i<8;++i){
            atomicAdd(&o1s[br][rb+i][u][0], acc[i]*Yv[i][0]);
            atomicAdd(&o1s[br][rb+i][u][1], acc[i]*Yv[i][1]);
            atomicAdd(&o1s[br][rb+i][u][2], acc[i]*Yv[i][2]);
          }
        } else if (n < 144){               // R10: o0 += T*Y[m] (1/sqrt3 folded in E)
          int q=n-48; int u=q/3, m=q%3;
#pragma unroll
          for(int i=0;i<8;++i) atomicAdd(&o0s[br][rb+i][u], acc[i]*Yv[i][m]);
        } else if (n < 192){               // R110: o1[u][m] += T
          int q=n-144; int u=q/3, m=q%3;
#pragma unroll
          for(int i=0;i<8;++i) atomicAdd(&o1s[br][rb+i][u][m], acc[i]);
        } else {                           // R111: o1[u] += (t x Y)  (1/sqrt2 folded in E)
          int q=n-192; int u=q/3, ii=q%3;
          int i1=(ii+1)%3, i2=(ii+2)%3;
#pragma unroll
          for(int i=0;i<8;++i){
            atomicAdd(&o1s[br][rb+i][u][i1], -acc[i]*Yv[i][i2]);
            atomicAdd(&o1s[br][rb+i][u][i2],  acc[i]*Yv[i][i1]);
          }
        }
      }
    }
  }
  __syncthreads();

  // flush block partials to global accumulators
  for(int i=tid;i<1024;i+=256){
    int br=i>>9; int al=(i>>5)&15; int u=i&31;
    atomicAdd(o0g + ((size_t)((z*NATOMS + a0 + al)*2 + br))*32 + u, o0s[br][al][u]);
  }
  for(int i=tid;i<1536;i+=256){
    int br=i/768; int r=i%768; int al=r/48; int q=r%48;
    atomicAdd(o1g + ((size_t)((z*NATOMS + a0 + al)*2 + br))*48 + q, (&o1s[br][al][0][0])[q]);
  }
}

// ---------------- gate: nonlinearity + write next-layer g (layer1) or features (layer2) --
__global__ void gate_kernel(const float* __restrict__ o0g, const float* __restrict__ o1g,
                            const float* __restrict__ mask, int layer,
                            float* __restrict__ g0out, float* __restrict__ g1out,
                            float* __restrict__ fout){
  int t = blockIdx.x*blockDim.x + threadIdx.x;  // == z*128+a
  if (t >= BZ*NATOMS) return;
  float mk = mask[t];
#pragma unroll
  for(int br=0;br<2;++br){
    const float* o0 = o0g + (size_t)(t*2+br)*32;
    const float* o1 = o1g + (size_t)(t*2+br)*48;
#pragma unroll
    for(int u=0;u<16;++u){
      float s  = sp_act(o0[u]) * mk;
      float gg = sigmoidf_(o0[16+u]);
      float v0 = o1[u*3+0]*gg*mk;
      float v1 = o1[u*3+1]*gg*mk;
      float v2 = o1[u*3+2]*gg*mk;
      if (layer==1){
        float sc = INV_SQRTN*mk;
        g0out[(t*2+br)*16 + u] = s*sc;
        float* gp = g1out + (t*2+br)*48 + u*3;
        gp[0]=v0*sc; gp[1]=v1*sc; gp[2]=v2*sc;
      } else {
        float* fp = fout + (size_t)t*128 + br*64;
        fp[u] = s;
        fp[16 + u*3 + 0] = v0;
        fp[16 + u*3 + 1] = v1;
        fp[16 + u*3 + 2] = v2;
      }
    }
  }
}

// ---------------- K5: FC head + per-atom batchnorms + masked sum ------------------------
__global__ void head_kernel(const float* __restrict__ f, const float* __restrict__ W1,
                            const float* __restrict__ B1, const float* __restrict__ W2,
                            const float* __restrict__ B2, const float* __restrict__ mask,
                            float* __restrict__ ybuf, float* __restrict__ out){
  __shared__ float red[2][NATOMS];
  int tid = threadIdx.x;
  for(int idx=tid; idx<BZ*NATOMS*128; idx+=256){
    int c = idx & 127; int zn = idx >> 7;
    const float* fr = f + (size_t)zn*128;
    float s = B1[c];
    for(int k=0;k<128;++k) s += fr[k]*W1[k*128+c];
    ybuf[idx] = s;
  }
  __threadfence();
  __syncthreads();
  if (tid < NATOMS){
    int n = tid;
    float m = 0.f;
    for(int z=0;z<2;++z){ const float* yr = ybuf + (size_t)(z*NATOMS+n)*128; for(int c=0;c<128;++c) m += yr[c]; }
    m *= (1.0f/256.0f);
    float vv = 0.f;
    for(int z=0;z<2;++z){ const float* yr = ybuf + (size_t)(z*NATOMS+n)*128; for(int c=0;c<128;++c){ float d=yr[c]-m; vv += d*d; } }
    vv *= (1.0f/256.0f);
    float rstd = rsqrtf(vv + 1e-5f);
    float y2[2];
    for(int z=0;z<2;++z){
      const float* yr = ybuf + (size_t)(z*NATOMS+n)*128;
      float s = B2[0];
      for(int c=0;c<128;++c){
        float t2 = (yr[c]-m)*rstd;
        t2 = (t2<0.f)? 0.2f*t2 : t2;
        s += t2*W2[c];
      }
      y2[z]=s;
    }
    float m2 = 0.5f*(y2[0]+y2[1]);
    float d0=y2[0]-m2, d1=y2[1]-m2;
    float v2 = 0.5f*(d0*d0+d1*d1);
    float rs2 = rsqrtf(v2+1e-5f);
    for(int z=0;z<2;++z){
      float t2 = (y2[z]-m2)*rs2;
      t2 = (t2<0.f)? 0.2f*t2 : t2;
      red[z][n] = t2 * mask[z*NATOMS+n];
    }
  }
  __syncthreads();
  if (tid < 2){
    float s=0.f;
    for(int n=0;n<NATOMS;++n) s += red[tid][n];
    out[tid]=s;
  }
}

extern "C" void kernel_launch(void* const* d_in, const int* in_sizes, int n_in,
                              void* d_out, int out_size, void* d_ws, size_t ws_size,
                              hipStream_t stream){
  const float* features  = (const float*)d_in[0];
  const float* geometry  = (const float*)d_in[1];
  const float* mask      = (const float*)d_in[2];
  const float* W_enc_bio = (const float*)d_in[3];
  const float* b_enc_bio = (const float*)d_in[4];
  const float* W_enc_chg = (const float*)d_in[5];
  const float* b_enc_chg = (const float*)d_in[6];
  const float* rad1_W0   = (const float*)d_in[7];
  const float* rad1_b0   = (const float*)d_in[8];
  const float* rad1_W1   = (const float*)d_in[9];
  const float* rad1_b1   = (const float*)d_in[10];
  const float* rad1_Wout = (const float*)d_in[11];
  const float* rad1_bout = (const float*)d_in[12];
  const float* rad2_W0   = (const float*)d_in[13];
  const float* rad2_b0   = (const float*)d_in[14];
  const float* rad2_W1   = (const float*)d_in[15];
  const float* rad2_b1   = (const float*)d_in[16];
  const float* rad2_Wout = (const float*)d_in[17];
  const float* rad2_bout = (const float*)d_in[18];
  const float* fc1_W     = (const float*)d_in[19];
  const float* fc1_b     = (const float*)d_in[20];
  const float* fc2_W     = (const float*)d_in[21];
  const float* fc2_b     = (const float*)d_in[22];
  float* out = (float*)d_out;

  char* ws = (char*)d_ws;
  size_t off = 0;
  auto take = [&](size_t bytes)->void*{
    off = (off + 255) & ~(size_t)255;
    void* p = ws + off; off += bytes; return p;
  };
  __bf16* basis = (__bf16*)take((size_t)BZ*NATOMS*NATOMS*16*2);
  float*  Y1    = (float*) take((size_t)BZ*NATOMS*NATOMS*3*4);
  __bf16* h1    = (__bf16*)take((size_t)BZ*NATOMS*NATOMS*KH*2);
  __bf16* h2    = (__bf16*)take((size_t)BZ*NATOMS*NATOMS*KH*2);
  float*  g0l1  = (float*) take((size_t)BZ*NATOMS*2*16*4);
  __bf16* E1    = (__bf16*)take((size_t)BZ*NATOMS*96*KH*2);
  float*  g0l2  = (float*) take((size_t)BZ*NATOMS*2*16*4);
  float*  g1l2  = (float*) take((size_t)BZ*NATOMS*2*48*4);
  __bf16* E2    = (__bf16*)take((size_t)BZ*NATOMS*480*KH*2);
  float*  fbuf  = (float*) take((size_t)BZ*NATOMS*128*4);
  float*  ybuf  = (float*) take((size_t)BZ*NATOMS*128*4);
  float*  o0g   = (float*) take((size_t)BZ*NATOMS*2*32*4);   // contiguous with o1g
  float*  o1g   = (float*) take((size_t)BZ*NATOMS*2*48*4);
  (void)ws_size; (void)n_in; (void)in_sizes; (void)out_size;
  const int ON = BZ*NATOMS*2*(32+48);   // floats in o0g+o1g (contiguous, both 256B-multiples)

  encode_kernel<<<1,256,0,stream>>>(features, mask, W_enc_bio, b_enc_bio, W_enc_chg, b_enc_chg, g0l1);
  geom_kernel<<<128,256,0,stream>>>(geometry, basis, Y1);
  radial_kernel<<<512,128,0,stream>>>(basis, rad1_W0, rad1_b0, rad1_W1, rad1_b1, h1);
  radial_kernel<<<512,128,0,stream>>>(basis, rad2_W0, rad2_b0, rad2_W1, rad2_b1, h2);

  e1_build_kernel<<<96,256,0,stream>>>(g0l1, rad1_Wout, rad1_bout, E1);
  zero_kernel<<<(ON+255)/256,256,0,stream>>>(o0g, ON);
  contract_accum_kernel<96,1><<<128,256,0,stream>>>(h1, E1, Y1, o0g, o1g);
  gate_kernel<<<1,256,0,stream>>>(o0g, o1g, mask, 1, g0l2, g1l2, nullptr);

  e2_build_kernel<<<480,256,0,stream>>>(g0l2, g1l2, rad2_Wout, rad2_bout, E2);
  zero_kernel<<<(ON+255)/256,256,0,stream>>>(o0g, ON);
  contract_accum_kernel<480,2><<<128,256,0,stream>>>(h2, E2, Y1, o0g, o1g);
  gate_kernel<<<1,256,0,stream>>>(o0g, o1g, mask, 2, nullptr, nullptr, fbuf);

  head_kernel<<<1,256,0,stream>>>(fbuf, fc1_W, fc1_b, fc2_W, fc2_b, mask, ybuf, out);
}
